// MultiHeadAttention_240518169076
// MI455X (gfx1250) — matrix operand order repaired and checked by reading, compiled, not run on hardware
//
#include <hip/hip_runtime.h>
#include <hip/hip_bf16.h>
#include <stdint.h>
#include <stddef.h>

// ---- CDNA5 feature detection -----------------------------------------------
#if defined(__has_builtin)
# if __has_builtin(__builtin_amdgcn_tensor_load_to_lds)
#  define HAVE_TDM 1
# else
#  define HAVE_TDM 0
# endif
#else
# define HAVE_TDM 0
#endif

#if HAVE_TDM
# if __has_include(<hip/amd_detail/amd_gfx1250_TDM.h>)
#  define TDM_6ARG 1            // therock-10.0 headers -> 6-arg builtin
# else
#  define TDM_6ARG 0            // ROCm 7.2 -> 5-arg builtin
# endif
#endif

namespace {

constexpr int D  = 1024;
constexpr int H  = 16;
constexpr int DK = 64;
constexpr int S  = 2048;
constexpr int B  = 4;
constexpr int BS = B * S;   // 8192

typedef __attribute__((ext_vector_type(16))) __bf16 v16bf;
typedef __attribute__((ext_vector_type(8)))  float  v8f;
typedef unsigned int v4u __attribute__((ext_vector_type(4)));
typedef int          v8i __attribute__((ext_vector_type(8)));
typedef int          v4i __attribute__((ext_vector_type(4)));

union Frag {
  v16bf v;
  uint4 q[2];
};

__device__ __forceinline__ __bf16 f2bf(float f) {
  union { float f; uint32_t u; } a; a.f = f;
  uint32_t r = a.u + 0x7fffu + ((a.u >> 16) & 1u);   // round-to-nearest-even
  union { uint16_t u; __bf16 h; } o; o.u = (uint16_t)(r >> 16);
  return o.h;
}

// ---- async global->LDS copy (16B per lane), ISA §10.7 ----------------------
__device__ __forceinline__ void async_cp16(void* lds_dst, const void* gsrc) {
  asm volatile("global_load_async_to_lds_b128 %0, %1, off"
               :: "v"((uint32_t)(uintptr_t)lds_dst),
                  "v"((uint64_t)(uintptr_t)gsrc)
               : "memory");
}
__device__ __forceinline__ void async_wait() {
  asm volatile("s_wait_asynccnt 0x0" ::: "memory");
}
__device__ __forceinline__ void tensor_wait() {
#if defined(__has_builtin) && __has_builtin(__builtin_amdgcn_s_wait_tensorcnt)
  __builtin_amdgcn_s_wait_tensorcnt(0);
#else
  asm volatile("s_wait_tensorcnt 0x0" ::: "memory");
#endif
}

// ---- TDM 2-D tile load: tile==tensor window, LDS row padding via D# --------
// w0pad packs G1 word0: data_size=2B | pad_enable | pad_interval | pad_amount.
__device__ __forceinline__ void tdm_load_2d(uint32_t lds_off, const void* g,
                                            int w0pad, int dim0, int rows,
                                            int stride) {
#if HAVE_TDM
  uint64_t ga = (uint64_t)(uintptr_t)g;
  v4u g0 = { 1u,                                  // count=1, user mode
             lds_off,                             // D#.lds_addr (bytes)
             (uint32_t)ga,                        // global_addr[31:0]
             (uint32_t)(ga >> 32) | 0x80000000u };// global_addr[56:32] | type=2
  v8i g1 = { w0pad,                               // mask=0, data_size=1(2B), pad cfg
             (dim0 & 0xffff) << 16,               // tensor_dim0[15:0]
             (rows & 0xffff) << 16,               // tensor_dim0[31:16]=0 | tensor_dim1[15:0]
             ((rows >> 16) & 0xffff) | ((dim0 & 0xffff) << 16), // dim1 hi | tile_dim0
             rows & 0xffff,                       // tile_dim1 | tile_dim2=0
             stride,                              // tensor_dim0_stride[31:0]
             0, 0 };
  v4i z4 = { 0, 0, 0, 0 };
# if TDM_6ARG
  v8i z8 = { 0, 0, 0, 0, 0, 0, 0, 0 };
  __builtin_amdgcn_tensor_load_to_lds(g0, g1, z4, z4, z8, 0);
# else
  __builtin_amdgcn_tensor_load_to_lds(g0, g1, z4, z4, 0);
# endif
#else
  (void)lds_off; (void)g; (void)w0pad; (void)dim0; (void)rows; (void)stride;
#endif
}

// G1.w0 for 32-elem rows (16 dwords): pad_interval code 3 (=16dw), pad 4 dwords.
constexpr int TDM_W0_32 = 0x00010000 | (1 << 20) | (3 << 22) | (3 << 25);

__device__ __forceinline__ float rmax16(float v) {
  v = fmaxf(v, __shfl_xor(v, 1, 32));
  v = fmaxf(v, __shfl_xor(v, 2, 32));
  v = fmaxf(v, __shfl_xor(v, 4, 32));
  v = fmaxf(v, __shfl_xor(v, 8, 32));
  return v;
}
__device__ __forceinline__ float rsum16(float v) {
  v += __shfl_xor(v, 1, 32);
  v += __shfl_xor(v, 2, 32);
  v += __shfl_xor(v, 4, 32);
  v += __shfl_xor(v, 8, 32);
  return v;
}

// ---------------- fp32 -> bf16 bulk convert (4 elems/thread) ----------------
__global__ void cvt_kernel(const float* __restrict__ in,
                           __bf16* __restrict__ out, int n) {
  int i = (blockIdx.x * blockDim.x + threadIdx.x) * 4;
  if (i < n) {
    float4 f = *(const float4*)(in + i);
    union { __bf16 h[4]; uint2 u; } o;
    o.h[0] = f2bf(f.x); o.h[1] = f2bf(f.y);
    o.h[2] = f2bf(f.z); o.h[3] = f2bf(f.w);
    *(uint2*)(out + i) = o.u;
  }
}

// --------------------------- projection GEMM --------------------------------
// y[m,n] = sum_k A[m,k]*W[n,k] + bias[n]   (torch Linear: x @ W^T + b)
// 256 threads = 8 waves (4 along M x 2 along N); wave 32x32 -> block 128x64.
// Double-buffered K loop: TDM (or async-to-LDS) stages tile k+1 under the WMMAs.
template <bool OUT_F32>
__global__ __launch_bounds__(256)
void gemm_kernel(const __bf16* __restrict__ A, const __bf16* __restrict__ W,
                 const float* __restrict__ bias, __bf16* __restrict__ outB,
                 float* __restrict__ outF) {
  constexpr int LDA = 40;              // padded LDS stride (elems); matches TDM pad
  __shared__ __bf16 sA[2][128 * LDA];
  __shared__ __bf16 sB[2][64 * LDA];

  const int t    = threadIdx.x;
  const int lane = t & 31;
  const int wave = t >> 5;
  const int wm   = wave & 3;
  const int wn   = wave >> 2;
  const int half = lane >> 4;
  const int lrow = lane & 15;

  const int m0 = blockIdx.x * 128;
  const int n0 = blockIdx.y * 64;

  auto stage = [&](int k0, int buf) {
    if (HAVE_TDM) {
      if (wave == 0) {
        tdm_load_2d((uint32_t)(uintptr_t)&sA[buf][0],
                    A + (size_t)m0 * D + k0, TDM_W0_32, 32, 128, D);
        tdm_load_2d((uint32_t)(uintptr_t)&sB[buf][0],
                    W + (size_t)n0 * D + k0, TDM_W0_32, 32, 64, D);
      }
    } else {
      int c = t;                           // A: 512 16B chunks, W: 256 chunks
      #pragma unroll
      for (int it = 0; it < 2; ++it) {
        int row = c >> 2, c8 = c & 3;
        async_cp16(&sA[buf][row * LDA + c8 * 8],
                   A + (size_t)(m0 + row) * D + k0 + c8 * 8);
        c += 256;
      }
      int row = t >> 2, c8 = t & 3;
      async_cp16(&sB[buf][row * LDA + c8 * 8],
                 W + (size_t)(n0 + row) * D + k0 + c8 * 8);
    }
  };
  auto stage_wait = [&]() {
    if (HAVE_TDM) { if (wave == 0) tensor_wait(); }
    else async_wait();
  };

  v8f acc[2][2] = {};

  stage(0, 0);
  stage_wait();
  __syncthreads();

  int buf = 0;
  for (int k0 = 0; k0 < D; k0 += 32, buf ^= 1) {
    if (k0 + 32 < D) stage(k0 + 32, buf ^ 1);   // DMA next tile under compute

    Frag a[2], b[2];
    #pragma unroll
    for (int mi = 0; mi < 2; ++mi) {   // A frag: lane = row M, K split by half
      int ar = wm * 32 + mi * 16 + lrow;
      a[mi].q[0] = *(const uint4*)(&sA[buf][ar * LDA + 8 * half]);
      a[mi].q[1] = *(const uint4*)(&sA[buf][ar * LDA + 16 + 8 * half]);
    }
    #pragma unroll
    for (int ni = 0; ni < 2; ++ni) {   // B frag: lane = col n, K split by half
      int br = wn * 32 + ni * 16 + lrow;
      b[ni].q[0] = *(const uint4*)(&sB[buf][br * LDA + 8 * half]);
      b[ni].q[1] = *(const uint4*)(&sB[buf][br * LDA + 16 + 8 * half]);
    }
    #pragma unroll
    for (int mi = 0; mi < 2; ++mi)
      #pragma unroll
      for (int ni = 0; ni < 2; ++ni)
        acc[mi][ni] = __builtin_amdgcn_wmma_f32_16x16x32_bf16(
            false, a[mi].v, false, b[ni].v, (short)0, acc[mi][ni], false, false);

    stage_wait();
    __syncthreads();
  }

  #pragma unroll
  for (int mi = 0; mi < 2; ++mi) {
    #pragma unroll
    for (int ni = 0; ni < 2; ++ni) {
      int n_g = n0 + wn * 32 + ni * 16 + lrow;
      float bv = bias[n_g];
      #pragma unroll
      for (int r = 0; r < 8; ++r) {     // D layout: N=lrow, M=r+8*half
        int m_g = m0 + wm * 32 + mi * 16 + r + 8 * half;
        float v = acc[mi][ni][r] + bv;
        if (OUT_F32) outF[(size_t)m_g * D + n_g] = v;
        else         outB[(size_t)m_g * D + n_g] = f2bf(v);
      }
    }
  }
}

// --------------------------- flash attention --------------------------------
// One block per (band of 256 Q rows, b*h). 8 waves, 32 Q rows per wave.
// KV streamed in 32-wide chunks; K async-DMAed row-major, V transposed in LDS.
__global__ __launch_bounds__(256)
void attn_kernel(const __bf16* __restrict__ Q, const __bf16* __restrict__ K,
                 const __bf16* __restrict__ V, __bf16* __restrict__ O) {
  constexpr int LK = 72;  // 144B rows (16B multiple)
  constexpr int LV = 40;  // 80B rows
  constexpr int LP = 40;
  __shared__ __bf16 sK[32 * LK];
  __shared__ __bf16 sVt[64 * LV];
  __shared__ __bf16 sP[8][32 * LP];

  const int t    = threadIdx.x;
  const int lane = t & 31;
  const int wave = t >> 5;
  const int half = lane >> 4;
  const int lrow = lane & 15;

  const int band = blockIdx.x;   // 0..7
  const int bh   = blockIdx.y;   // 0..63
  const int b    = bh >> 4;
  const int h    = bh & 15;

  const __bf16* qb = Q + ((size_t)b * S) * D + h * DK;
  const __bf16* kb = K + ((size_t)b * S) * D + h * DK;
  const __bf16* vb = V + ((size_t)b * S) * D + h * DK;

  const int q0 = band * 256 + wave * 32;

  // resident Q fragments: [mi][ks], DK=64 = 2 K-steps of 32
  Frag aq[2][2];
  #pragma unroll
  for (int mi = 0; mi < 2; ++mi)
    #pragma unroll
    for (int ks = 0; ks < 2; ++ks) {
      const __bf16* qr = qb + (size_t)(q0 + mi * 16 + lrow) * D + ks * 32;
      aq[mi][ks].q[0] = *(const uint4*)(qr + 8 * half);
      aq[mi][ks].q[1] = *(const uint4*)(qr + 16 + 8 * half);
    }

  v8f accO[2][4] = {};
  float rowm[2][8], rowl[2][8];
  #pragma unroll
  for (int mi = 0; mi < 2; ++mi)
    #pragma unroll
    for (int r = 0; r < 8; ++r) { rowm[mi][r] = -3.0e30f; rowl[mi][r] = 0.f; }

  const float scale = 0.125f;           // 1/sqrt(DK)
  const int Jmax = band * 8 + 8;        // causal: later chunks fully masked

  for (int j = 0; j < Jmax; ++j) {
    {  // stage K chunk via async-to-LDS; V chunk transposed via VGPRs
      int row = t >> 3, c8 = t & 7;     // 32 rows x 8 chunks of 8
      async_cp16(sK + row * LK + c8 * 8,
                 kb + (size_t)(j * 32 + row) * D + c8 * 8);
      union { uint4 q; __bf16 h[8]; } dv;
      dv.q = *(const uint4*)(vb + (size_t)(j * 32 + row) * D + c8 * 8);
      #pragma unroll
      for (int e = 0; e < 8; ++e)
        sVt[(c8 * 8 + e) * LV + row] = dv.h[e];
    }
    async_wait();
    __syncthreads();

    if (j <= band * 8 + wave) {         // per-wave causal chunk filter
      // ---- S = Q K^T ----
      v8f accS[2][2] = {};
      #pragma unroll
      for (int ks = 0; ks < 2; ++ks) {
        Frag bk[2];
        #pragma unroll
        for (int ni = 0; ni < 2; ++ni) {
          int br = ni * 16 + lrow;
          bk[ni].q[0] = *(const uint4*)(sK + br * LK + ks * 32 + 8 * half);
          bk[ni].q[1] = *(const uint4*)(sK + br * LK + ks * 32 + 16 + 8 * half);
        }
        #pragma unroll
        for (int mi = 0; mi < 2; ++mi)
          #pragma unroll
          for (int ni = 0; ni < 2; ++ni)
            accS[mi][ni] = __builtin_amdgcn_wmma_f32_16x16x32_bf16(
                false, aq[mi][ks].v, false, bk[ni].v, (short)0,
                accS[mi][ni], false, false);
      }

      // ---- online softmax (row = r + 8*half per C-layout) ----
      #pragma unroll
      for (int mi = 0; mi < 2; ++mi) {
        #pragma unroll
        for (int r = 0; r < 8; ++r) {
          int rg = q0 + mi * 16 + r + 8 * half;
          float s0 = accS[mi][0][r] * scale;
          float s1 = accS[mi][1][r] * scale;
          int c0 = j * 32 + lrow;
          int c1 = j * 32 + 16 + lrow;
          if (c0 > rg) s0 = -3.0e30f;
          if (c1 > rg) s1 = -3.0e30f;
          float mnew = fmaxf(rowm[mi][r], rmax16(fmaxf(s0, s1)));
          float f  = __expf(rowm[mi][r] - mnew);
          float p0 = __expf(s0 - mnew);
          float p1 = __expf(s1 - mnew);
          rowl[mi][r] = rowl[mi][r] * f + rsum16(p0 + p1);
          rowm[mi][r] = mnew;
          #pragma unroll
          for (int di = 0; di < 4; ++di) accO[mi][di][r] *= f;
          int pr = mi * 16 + r + 8 * half;          // re-layout P via wave LDS scratch
          sP[wave][pr * LP + lrow]      = f2bf(p0);
          sP[wave][pr * LP + 16 + lrow] = f2bf(p1);
        }
      }

      // ---- O += P V  (B frags from transposed V: contiguous reads) ----
      Frag bv[4];
      #pragma unroll
      for (int di = 0; di < 4; ++di) {
        int vr = di * 16 + lrow;
        bv[di].q[0] = *(const uint4*)(sVt + vr * LV + 8 * half);
        bv[di].q[1] = *(const uint4*)(sVt + vr * LV + 16 + 8 * half);
      }
      #pragma unroll
      for (int mi = 0; mi < 2; ++mi) {
        Frag ap;
        int prr = mi * 16 + lrow;
        ap.q[0] = *(const uint4*)(sP[wave] + prr * LP + 8 * half);
        ap.q[1] = *(const uint4*)(sP[wave] + prr * LP + 16 + 8 * half);
        #pragma unroll
        for (int di = 0; di < 4; ++di)
          accO[mi][di] = __builtin_amdgcn_wmma_f32_16x16x32_bf16(
              false, ap.v, false, bv[di].v, (short)0, accO[mi][di], false, false);
      }
    }
    __syncthreads();
  }

  // ---- normalize + store to [B,S,D] bf16 ----
  #pragma unroll
  for (int mi = 0; mi < 2; ++mi) {
    #pragma unroll
    for (int r = 0; r < 8; ++r) {
      float inv = 1.0f / rowl[mi][r];
      int m_g = b * S + q0 + mi * 16 + r + 8 * half;
      #pragma unroll
      for (int di = 0; di < 4; ++di) {
        int n_g = h * DK + di * 16 + lrow;
        O[(size_t)m_g * D + n_g] = f2bf(accO[mi][di][r] * inv);
      }
    }
  }
}

}  // namespace

extern "C" void kernel_launch(void* const* d_in, const int* in_sizes, int n_in,
                              void* d_out, int out_size, void* d_ws, size_t ws_size,
                              hipStream_t stream) {
  (void)in_sizes; (void)n_in; (void)out_size; (void)ws_size;
  const float* x  = (const float*)d_in[0];
  const float* wq = (const float*)d_in[1];
  const float* bq = (const float*)d_in[2];
  const float* wk = (const float*)d_in[3];
  const float* bk = (const float*)d_in[4];
  const float* wv = (const float*)d_in[5];
  const float* bv = (const float*)d_in[6];
  const float* wo = (const float*)d_in[7];
  const float* bo = (const float*)d_in[8];
  float* out = (float*)d_out;

  // workspace: x_bf 16MB | 4 weights 2MB each | q,k,v 16MB each | attn_out 16MB
  char* ws = (char*)d_ws;
  __bf16* xb  = (__bf16*)(ws + 0);
  __bf16* wqb = (__bf16*)(ws + 16777216);
  __bf16* wkb = (__bf16*)(ws + 18874368);
  __bf16* wvb = (__bf16*)(ws + 20971520);
  __bf16* wob = (__bf16*)(ws + 23068672);
  __bf16* qb  = (__bf16*)(ws + 25165824);
  __bf16* kb  = (__bf16*)(ws + 41943040);
  __bf16* vb  = (__bf16*)(ws + 58720256);
  __bf16* aob = (__bf16*)(ws + 75497472);   // total 88 MB

  cvt_kernel<<<dim3(BS * D / 4 / 256), 256, 0, stream>>>(x, xb, BS * D);
  cvt_kernel<<<dim3(D * D / 4 / 256), 256, 0, stream>>>(wq, wqb, D * D);
  cvt_kernel<<<dim3(D * D / 4 / 256), 256, 0, stream>>>(wk, wkb, D * D);
  cvt_kernel<<<dim3(D * D / 4 / 256), 256, 0, stream>>>(wv, wvb, D * D);
  cvt_kernel<<<dim3(D * D / 4 / 256), 256, 0, stream>>>(wo, wob, D * D);

  dim3 gg(BS / 128, D / 64);
  gemm_kernel<false><<<gg, 256, 0, stream>>>(xb, wqb, bq, qb, nullptr);
  gemm_kernel<false><<<gg, 256, 0, stream>>>(xb, wkb, bk, kb, nullptr);
  gemm_kernel<false><<<gg, 256, 0, stream>>>(xb, wvb, bv, vb, nullptr);

  attn_kernel<<<dim3(8, 64), 256, 0, stream>>>(qb, kb, vb, aob);

  gemm_kernel<true><<<gg, 256, 0, stream>>>(aob, wob, bo, nullptr, out);
}
